// Qwen35Attention_64269890618113
// MI455X (gfx1250) — compile-verified, matrix operand order checked by reading, not run on hardware
//
#include <hip/hip_runtime.h>
#include <hip/hip_bf16.h>

#define B_   2
#define S_   2048
#define D_   2048
#define KVH_ 4
#define QPG_ 4
#define HD_  128

typedef __attribute__((ext_vector_type(16))) __bf16 bf16x16;
typedef __attribute__((ext_vector_type(8)))  float  f32x8;
typedef __attribute__((ext_vector_type(4)))  int    i32x4;
typedef __attribute__((ext_vector_type(4)))  float  f32x4;

union FragU { bf16x16 v; i32x4 q[2]; };

__device__ __forceinline__ unsigned short f2bf(float f) {
  union { float f; unsigned u; } c; c.f = f;
  unsigned u = c.u;
  return (unsigned short)((u + 0x7FFFu + ((u >> 16) & 1u)) >> 16);
}

// Async global->LDS 16B-per-lane copy (CDNA5 GLOBAL_LOAD_ASYNC_TO_LDS_B128,
// tracked by ASYNCcnt). lds_off is the addrspace(3) byte offset (low 32 bits
// of the generic pointer), g is the per-lane 64-bit global address.
__device__ __forceinline__ void async_ld128(unsigned lds_off, const void* g) {
  asm volatile("global_load_async_to_lds_b128 %0, %1, off"
               :: "v"(lds_off), "v"(g) : "memory");
}
__device__ __forceinline__ void wait_async0() {
  asm volatile("s_wait_asynccnt 0x0" ::: "memory");
}

// Load a 16x32 bf16 fragment (A-layout / B^T-layout) from LDS.
// lane 0-15: row=lane, K chunks [0..7] and [16..23]
// lane 16-31: row=lane-16, K chunks [8..15] and [24..31]
__device__ __forceinline__ bf16x16 ldfrag(const unsigned short* p0, int stride, int lane) {
  const int row = lane & 15;
  const int lh  = (lane >> 4) << 3;   // 0 or 8
  const unsigned short* p = p0 + row * stride + lh;
  FragU u;
  u.q[0] = *(const i32x4*)(p);
  u.q[1] = *(const i32x4*)(p + 16);
  return u.v;
}

// ---------------------------------------------------------------------------
// Generic bf16 WMMA GEMM: C[M,N] = A[M,K] * B[N,K]^T   (A,B bf16, C f32)
// 128x128 block tile, 8 waves (4M x 2N), each wave 32x64 = 2x4 WMMA tiles.
// Double-buffered LDS filled by async global->LDS copies: tile k+1 streams in
// while tile k is consumed by WMMA.
// ---------------------------------------------------------------------------
__global__ __launch_bounds__(256) void gemm_bf16_wmma(
    const unsigned short* __restrict__ A, const unsigned short* __restrict__ Bw,
    float* __restrict__ C, int M, int N, int K)
{
  constexpr int LS = 40;  // 32 + 8 pad (16B aligned rows, dodges bank conflicts)
  __shared__ __align__(16) unsigned short As[2][128 * LS];
  __shared__ __align__(16) unsigned short Bs[2][128 * LS];

  const int tid  = threadIdx.x;
  const int lane = tid & 31;
  const int wave = tid >> 5;
  const int wm   = wave & 3;      // 0..3 -> 32-row slice
  const int wn   = wave >> 2;     // 0..1 -> 64-col slice
  const int nb   = N >> 7;
  const long mBase = (long)((int)blockIdx.x / nb) << 7;
  const long nBase = (long)((int)blockIdx.x % nb) << 7;

  f32x8 acc[2][4] = {};

  // async staging: 512 16B chunks per 128x32 tile, 2 per thread per matrix.
  // chunk c: row = c>>2 (4 chunks/row), col = (c&3)*8 elements
  const int c0 = tid;
  const int c1 = tid + 256;
  const int r0 = c0 >> 2, q0 = (c0 & 3) << 3;
  const int r1 = c1 >> 2, q1 = (c1 & 3) << 3;
  const unsigned short* Ab = A  + mBase * (long)K;
  const unsigned short* Bb = Bw + nBase * (long)K;

  auto issue_tile = [&](int bufi, int k0) {
    async_ld128((unsigned)(uintptr_t)&As[bufi][r0 * LS + q0], Ab + (long)r0 * K + k0 + q0);
    async_ld128((unsigned)(uintptr_t)&As[bufi][r1 * LS + q1], Ab + (long)r1 * K + k0 + q1);
    async_ld128((unsigned)(uintptr_t)&Bs[bufi][r0 * LS + q0], Bb + (long)r0 * K + k0 + q0);
    async_ld128((unsigned)(uintptr_t)&Bs[bufi][r1 * LS + q1], Bb + (long)r1 * K + k0 + q1);
  };

  issue_tile(0, 0);
  int buf = 0;
  for (int k0 = 0; k0 < K; k0 += 32) {
    wait_async0();       // this wave's tile-k copies have landed in LDS
    __syncthreads();     // all waves landed; previous compute done (safe to refill)
    if (k0 + 32 < K) issue_tile(buf ^ 1, k0 + 32);  // overlap with compute below

    bf16x16 af[2], bfv[4];
#pragma unroll
    for (int i = 0; i < 2; ++i) af[i]  = ldfrag(&As[buf][(wm * 32 + i * 16) * LS], LS, lane);
#pragma unroll
    for (int j = 0; j < 4; ++j) bfv[j] = ldfrag(&Bs[buf][(wn * 64 + j * 16) * LS], LS, lane);
#pragma unroll
    for (int i = 0; i < 2; ++i)
#pragma unroll
      for (int j = 0; j < 4; ++j)
        acc[i][j] = __builtin_amdgcn_wmma_f32_16x16x32_bf16(
            false, af[i], false, bfv[j], (short)0, acc[i][j], false, false);
    buf ^= 1;
  }

  const int cn = lane & 15;
  const int cm = (lane >> 4) << 3;
#pragma unroll
  for (int i = 0; i < 2; ++i)
#pragma unroll
    for (int j = 0; j < 4; ++j) {
      const long row = mBase + wm * 32 + i * 16 + cm;
      const long col = nBase + wn * 64 + j * 16 + cn;
#pragma unroll
      for (int r = 0; r < 8; ++r)
        C[(row + r) * (long)N + col] = acc[i][j][r];
    }
}

// ---------------------------------------------------------------------------
// fp32 -> bf16 cast (n multiple of 1024)
// ---------------------------------------------------------------------------
__global__ __launch_bounds__(256) void cast_f32_bf16(
    const float* __restrict__ in, unsigned short* __restrict__ out, long n)
{
  const long i = ((long)blockIdx.x * 256 + threadIdx.x) * 4;
  if (i >= n) return;
  f32x4 v = *(const f32x4*)(in + i);
#pragma unroll
  for (int j = 0; j < 4; ++j) out[i + j] = f2bf(v[j]);
}

// ---------------------------------------------------------------------------
// RMSNorm(1+w) + RoPE, output bf16. One wave per 128-wide head row.
// isQ: input row = qgraw[(b*S+s)*4096 + (g*4+p)*256 + o], scale=1/sqrt(128)
//      output [b][g][p][s][128]
// !isQ: input row = kraw[(b*S+s)*512 + g*128 + o], output [b][g][s][128]
// ---------------------------------------------------------------------------
__global__ __launch_bounds__(256) void normrope_kernel(
    const float* __restrict__ in, const float* __restrict__ w,
    unsigned short* __restrict__ out, int isQ)
{
  const int lane = threadIdx.x & 31;
  const int wv   = threadIdx.x >> 5;
  const int wid  = (int)blockIdx.x * 8 + wv;

  int s;
  size_t inoff;
  if (isQ) {
    s = wid & 2047;
    const int pp = (wid >> 11) & 3;
    const int gg = (wid >> 13) & 3;
    const int bb = wid >> 15;
    inoff = ((size_t)(bb * S_ + s)) * 4096 + (size_t)(gg * QPG_ + pp) * 256;
  } else {
    s = wid & 2047;
    const int gg = (wid >> 11) & 3;
    const int bb = wid >> 13;
    inoff = ((size_t)(bb * S_ + s)) * 512 + (size_t)gg * 128;
  }

  const int o0 = lane * 4;
  float h[4];
#pragma unroll
  for (int j = 0; j < 4; ++j) h[j] = in[inoff + o0 + j];

  float ss = h[0]*h[0] + h[1]*h[1] + h[2]*h[2] + h[3]*h[3];
#pragma unroll
  for (int m = 1; m < 32; m <<= 1) ss += __shfl_xor(ss, m, 32);
  const float rms = rsqrtf(ss * (1.0f / 128.0f) + 1e-6f);
#pragma unroll
  for (int j = 0; j < 4; ++j) h[j] = h[j] * rms * (1.0f + w[o0 + j]);

  const float pos   = (float)s;
  const float scale = isQ ? 0.08838834764831845f : 1.0f;  // 1/sqrt(HD) folded into Q
#pragma unroll
  for (int j = 0; j < 4; ++j) {
    const int oo = o0 + j;
    const float partner = __shfl_xor(h[j], 16, 32);       // element at oo ^ 64
    const float rot = (oo < 64) ? -partner : partner;
    const float f = pos * __powf(10000.0f, -(float)(oo & 63) * (1.0f / 64.0f));
    const float val = (h[j] * __cosf(f) + rot * __sinf(f)) * scale;
    out[(size_t)wid * 128 + oo] = f2bf(val);
  }
}

// ---------------------------------------------------------------------------
// V: f32 (B*S,512) -> bf16 [b][g][s][128]
// ---------------------------------------------------------------------------
__global__ __launch_bounds__(256) void vcast_kernel(
    const float* __restrict__ vraw, unsigned short* __restrict__ vb)
{
  const long i = ((long)blockIdx.x * 256 + threadIdx.x) * 4;
#pragma unroll
  for (int j = 0; j < 4; ++j) {
    const long idx = i + j;
    const int  h  = (int)(idx & 127);
    const int  s  = (int)((idx >> 7) & 2047);
    const int  gg = (int)((idx >> 18) & 3);
    const long bb = idx >> 20;
    vb[idx] = f2bf(vraw[((size_t)(bb * S_ + s)) * 512 + gg * 128 + h]);
  }
}

// ---------------------------------------------------------------------------
// Flash attention (causal, online softmax) + sigmoid gate.
// One block per (b,g,p, 128-row Q block); 8 waves, 16 Q rows each.
// QK^T and PV via v_wmma_f32_16x16x32_bf16; K tile streamed to LDS with
// async global->LDS copies; V^T staged via scalar LDS transpose.
// ---------------------------------------------------------------------------
__global__ __launch_bounds__(256) void flash_attn(
    const unsigned short* __restrict__ qb, const unsigned short* __restrict__ kb,
    const unsigned short* __restrict__ vb, const float* __restrict__ qgraw,
    unsigned short* __restrict__ attnb)
{
  constexpr int KS = 136;  // K tile row stride (128 + 8)
  constexpr int VS = 72;   // V^T row stride (64 + 8)
  constexpr int PS = 72;   // P row stride
  __shared__ __align__(16) unsigned short Ks[64 * KS];
  __shared__ __align__(16) unsigned short Vt[128 * VS];
  __shared__ __align__(16) unsigned short Pw[8 * 16 * PS];

  const int tid  = threadIdx.x;
  const int lane = tid & 31;
  const int wave = tid >> 5;
  int bidx = blockIdx.x;
  const int qblk = bidx & 15; bidx >>= 4;
  const int p = bidx & 3;     bidx >>= 2;
  const int g = bidx & 3;     bidx >>= 2;
  const int b = bidx;

  const int qBase = qblk << 7;
  const int rowM  = (lane >> 4) << 3;  // 0 or 8 (+r)
  const int colN  = lane & 15;

  // Q fragments for this wave's 16 rows (scale already folded in)
  const unsigned short* qrow =
      qb + ((size_t)((b * KVH_ + g) * QPG_ + p) * S_ + qBase + wave * 16 + (lane & 15)) * HD_;
  bf16x16 qf[4];
  {
    const int lh = (lane >> 4) << 3;
#pragma unroll
    for (int kk = 0; kk < 4; ++kk) {
      FragU u;
      u.q[0] = *(const i32x4*)(qrow + kk * 32 + lh);
      u.q[1] = *(const i32x4*)(qrow + kk * 32 + 16 + lh);
      qf[kk] = u.v;
    }
  }

  float m_r[8], l_r[8];
  f32x8 o[8] = {};
#pragma unroll
  for (int r = 0; r < 8; ++r) { m_r[r] = -3.0e38f; l_r[r] = 0.0f; }

  const unsigned short* kh = kb + (size_t)(b * KVH_ + g) * S_ * HD_;
  const unsigned short* vh = vb + (size_t)(b * KVH_ + g) * S_ * HD_;

  const int nKT = (qblk << 1) + 2;  // 64-key tiles up to and including the diagonal
  for (int kt = 0; kt < nKT; ++kt) {
    __syncthreads();   // previous tile fully consumed before refill
    // K tile [64][128] -> LDS via async copies (4 x b128 per lane)
#pragma unroll
    for (int c = 0; c < 4; ++c) {
      const int cid  = tid + c * 256;         // 0..1023
      const int krow = cid >> 4;
      const int kcol = (cid & 15) << 3;
      async_ld128((unsigned)(uintptr_t)&Ks[krow * KS + kcol],
                  kh + (size_t)(kt * 64 + krow) * HD_ + kcol);
    }
    // V tile transposed [128][64] -> LDS
    for (int i = tid; i < 64 * 128; i += 256) {
      const int vkey = i >> 7;
      const int vhd  = i & 127;
      Vt[vhd * VS + vkey] = vh[(size_t)(kt * 64 + vkey) * HD_ + vhd];
    }
    if (kt + 1 < nKT) {   // prefetch next tiles toward L2 while we compute
      __builtin_prefetch((const void*)(kh + (size_t)(kt * 64 + 64 + (tid >> 2)) * HD_), 0, 2);
      __builtin_prefetch((const void*)(vh + (size_t)(kt * 64 + 64 + (tid >> 2)) * HD_), 0, 2);
    }
    wait_async0();
    __syncthreads();

    // scores: 4 x (16x16) tiles, K-dim 128 in 4 WMMA steps
    f32x8 s[4];
#pragma unroll
    for (int nt = 0; nt < 4; ++nt) {
      f32x8 a = {};
#pragma unroll
      for (int kk = 0; kk < 4; ++kk) {
        bf16x16 kf = ldfrag(&Ks[(nt * 16) * KS + kk * 32], KS, lane);
        a = __builtin_amdgcn_wmma_f32_16x16x32_bf16(
            false, qf[kk], false, kf, (short)0, a, false, false);
      }
      s[nt] = a;
    }

    // causal mask + running row max
    float rmax[8];
#pragma unroll
    for (int r = 0; r < 8; ++r) rmax[r] = -3.0e38f;
#pragma unroll
    for (int nt = 0; nt < 4; ++nt) {
      const int key = kt * 64 + nt * 16 + colN;
#pragma unroll
      for (int r = 0; r < 8; ++r) {
        const int qr = qBase + wave * 16 + rowM + r;
        float v = s[nt][r];
        v = (key > qr) ? -3.0e38f : v;
        s[nt][r] = v;
        rmax[r] = fmaxf(rmax[r], v);
      }
    }
#pragma unroll
    for (int r = 0; r < 8; ++r)
#pragma unroll
      for (int m = 1; m < 16; m <<= 1)
        rmax[r] = fmaxf(rmax[r], __shfl_xor(rmax[r], m, 32));

    float sc[8], rsum[8];
#pragma unroll
    for (int r = 0; r < 8; ++r) {
      const float mn = fmaxf(m_r[r], rmax[r]);
      sc[r]   = __expf(m_r[r] - mn);
      m_r[r]  = mn;
      rsum[r] = 0.0f;
    }
#pragma unroll
    for (int nt = 0; nt < 4; ++nt)
#pragma unroll
      for (int r = 0; r < 8; ++r) {
        const float pp = __expf(s[nt][r] - m_r[r]);
        s[nt][r] = pp;
        rsum[r] += pp;
      }
#pragma unroll
    for (int r = 0; r < 8; ++r) {
#pragma unroll
      for (int m = 1; m < 16; m <<= 1) rsum[r] += __shfl_xor(rsum[r], m, 32);
      l_r[r] = l_r[r] * sc[r] + rsum[r];
    }
#pragma unroll
    for (int ot = 0; ot < 8; ++ot)
#pragma unroll
      for (int r = 0; r < 8; ++r) o[ot][r] *= sc[r];

    // C-layout P -> bf16 -> wave-private LDS -> A-fragment layout
    unsigned short* pr = &Pw[wave * 16 * PS];
#pragma unroll
    for (int nt = 0; nt < 4; ++nt)
#pragma unroll
      for (int r = 0; r < 8; ++r)
        pr[(rowM + r) * PS + nt * 16 + colN] = f2bf(s[nt][r]);

    bf16x16 pf[2];
#pragma unroll
    for (int kk = 0; kk < 2; ++kk) pf[kk] = ldfrag(pr + kk * 32, PS, lane);

    // O += P * V  (8 hd tiles x 2 K-steps)
#pragma unroll
    for (int ot = 0; ot < 8; ++ot)
#pragma unroll
      for (int kk = 0; kk < 2; ++kk) {
        bf16x16 vf = ldfrag(&Vt[(ot * 16) * VS + kk * 32], VS, lane);
        o[ot] = __builtin_amdgcn_wmma_f32_16x16x32_bf16(
            false, pf[kk], false, vf, (short)0, o[ot], false, false);
      }
  }

  // epilogue: normalize, sigmoid gate, store bf16 [b][s][hp*128 + hd]
  const int hp = g * QPG_ + p;
#pragma unroll
  for (int ot = 0; ot < 8; ++ot)
#pragma unroll
    for (int r = 0; r < 8; ++r) {
      const int sIdx = qBase + wave * 16 + rowM + r;
      const int hd   = ot * 16 + colN;
      float val = o[ot][r] / l_r[r];
      const float gt = qgraw[((size_t)(b * S_ + sIdx)) * 4096 + hp * 256 + 128 + hd];
      val *= 1.0f / (1.0f + __expf(-gt));
      attnb[((size_t)(b * S_ + sIdx)) * 2048 + hp * 128 + hd] = f2bf(val);
    }
}

// ---------------------------------------------------------------------------
extern "C" void kernel_launch(void* const* d_in, const int* in_sizes, int n_in,
                              void* d_out, int out_size, void* d_ws, size_t ws_size,
                              hipStream_t stream) {
  (void)in_sizes; (void)n_in; (void)out_size; (void)ws_size;
  const float* x   = (const float*)d_in[0];
  const float* wq  = (const float*)d_in[1];
  const float* wk  = (const float*)d_in[2];
  const float* wv  = (const float*)d_in[3];
  const float* wo  = (const float*)d_in[4];
  const float* qnw = (const float*)d_in[5];
  const float* knw = (const float*)d_in[6];
  float* out = (float*)d_out;

  char* ws = (char*)d_ws;
  size_t off = 0;
  auto alloc = [&](size_t bytes) -> void* {
    void* pp = ws + off;
    off += (bytes + 255) & ~(size_t)255;
    return pp;
  };
  unsigned short* xb    = (unsigned short*)alloc(8388608ull * 2);   // x bf16
  unsigned short* wqb   = (unsigned short*)alloc(8388608ull * 2);   // wq bf16
  unsigned short* wkb   = (unsigned short*)alloc(1048576ull * 2);
  unsigned short* wvb   = (unsigned short*)alloc(1048576ull * 2);
  unsigned short* wob   = (unsigned short*)alloc(4194304ull * 2);
  float*          qgraw = (float*)alloc(16777216ull * 4);           // (B*S, 4096) q|gate
  float*          kraw  = (float*)alloc(2097152ull * 4);            // (B*S, 512)
  float*          vraw  = (float*)alloc(2097152ull * 4);
  unsigned short* qbuf  = (unsigned short*)alloc(8388608ull * 2);   // [b][g][p][s][128]
  unsigned short* kbuf  = (unsigned short*)alloc(2097152ull * 2);   // [b][g][s][128]
  unsigned short* vbuf  = (unsigned short*)alloc(2097152ull * 2);
  unsigned short* attnb = (unsigned short*)alloc(8388608ull * 2);   // (B*S, 2048)

  // 1. casts to bf16
  cast_f32_bf16<<<8388608 / 1024, 256, 0, stream>>>(x,  xb,  8388608);
  cast_f32_bf16<<<8388608 / 1024, 256, 0, stream>>>(wq, wqb, 8388608);
  cast_f32_bf16<<<1048576 / 1024, 256, 0, stream>>>(wk, wkb, 1048576);
  cast_f32_bf16<<<1048576 / 1024, 256, 0, stream>>>(wv, wvb, 1048576);
  cast_f32_bf16<<<4194304 / 1024, 256, 0, stream>>>(wo, wob, 4194304);

  // 2. projections (WMMA GEMM, async double-buffered)
  gemm_bf16_wmma<<<(4096 / 128) * (4096 / 128), 256, 0, stream>>>(xb, wqb, qgraw, 4096, 4096, 2048);
  gemm_bf16_wmma<<<(4096 / 128) * (512 / 128),  256, 0, stream>>>(xb, wkb, kraw,  4096, 512,  2048);
  gemm_bf16_wmma<<<(4096 / 128) * (512 / 128),  256, 0, stream>>>(xb, wvb, vraw,  4096, 512,  2048);

  // 3. norm + rope (+scale for Q), V relayout
  normrope_kernel<<<65536 / 8, 256, 0, stream>>>(qgraw, qnw, qbuf, 1);
  normrope_kernel<<<16384 / 8, 256, 0, stream>>>(kraw,  knw, kbuf, 0);
  vcast_kernel<<<2097152 / 1024, 256, 0, stream>>>(vraw, vbuf);

  // 4. flash attention + gate
  flash_attn<<<B_ * KVH_ * QPG_ * (S_ / 128), 256, 0, stream>>>(qbuf, kbuf, vbuf, qgraw, attnb);

  // 5. output projection -> d_out (f32)
  gemm_bf16_wmma<<<(4096 / 128) * (2048 / 128), 256, 0, stream>>>(attnb, wob, out, 4096, 2048, 2048);
}
